// LightGCNHG_19705309954505
// MI455X (gfx1250) — compile-verified, matrix operand order checked by reading, not run on hardware
//
#include <hip/hip_runtime.h>

#define N_TOTAL    150100
#define EMBED_DIM  64
#define NUM_USERS  100000
#define N_EDGES    3000000
#define BATCH      4096
#define NUM_LAYERS 3

typedef __attribute__((ext_vector_type(2)))  float    v2f;
typedef __attribute__((ext_vector_type(8)))  float    v8f;
typedef __attribute__((ext_vector_type(16))) _Float16 v16h;

// ---------------- zero a big buffer (float4 streaming stores) ----------------
__global__ void lg_zero(float4* __restrict__ p, int n4) {
  int i = blockIdx.x * blockDim.x + threadIdx.x;
  const int stride = gridDim.x * blockDim.x;
  for (; i < n4; i += stride) p[i] = make_float4(0.f, 0.f, 0.f, 0.f);
}

// ------------- init acc_small[8192][64] = node_emb at gathered rows ----------
__global__ void lg_gather_init(const float* __restrict__ emb, const int* __restrict__ uid,
                               const int* __restrict__ iid, float* __restrict__ acc) {
  const int t = blockIdx.x * blockDim.x + threadIdx.x;   // 0 .. 2*BATCH*64-1
  const int r = t >> 6, c = t & 63;
  const int n = (r < BATCH) ? uid[r] : (NUM_USERS + iid[r - BATCH]);
  acc[t] = emb[(size_t)n * EMBED_DIM + c];
}

// ------------- acc_small += layer embedding at gathered rows -----------------
__global__ void lg_gather_acc(const float* __restrict__ emb, const int* __restrict__ uid,
                              const int* __restrict__ iid, float* __restrict__ acc) {
  const int t = blockIdx.x * blockDim.x + threadIdx.x;
  const int r = t >> 6, c = t & 63;
  const int n = (r < BATCH) ? uid[r] : (NUM_USERS + iid[r - BATCH]);
  acc[t] += emb[(size_t)n * EMBED_DIM + c];
}

// ------------- SpMM scatter: one wave per edge, float2 per lane --------------
__global__ void lg_spmm_scatter(const float* __restrict__ emb, const float* __restrict__ val,
                                const int* __restrict__ rw, const int* __restrict__ cl,
                                float* __restrict__ out) {
  const int lane = threadIdx.x & 31;
  const int d    = lane << 1;                       // 2 dims per lane, 64 total
  int w          = (blockIdx.x * blockDim.x + threadIdx.x) >> 5;
  const int nw   = (gridDim.x * blockDim.x) >> 5;
  for (int e = w; e < N_EDGES; e += nw) {
    const int   r = __builtin_amdgcn_readfirstlane(rw[e]);   // scalarize bases
    const int   c = __builtin_amdgcn_readfirstlane(cl[e]);
    const float v = val[e];
    const float2 s = *reinterpret_cast<const float2*>(emb + (size_t)c * EMBED_DIM + d);
    float* dst = out + (size_t)r * EMBED_DIM + d;
    unsafeAtomicAdd(dst,     v * s.x);              // global_atomic_add_f32
    unsafeAtomicAdd(dst + 1, v * s.y);
  }
}

// ------------- batched dot products via f32 WMMA -----------------------------
// Wave w computes 16 pairs: D = U(16x64) x I^T(64x16), diagonal = dot products.
__global__ void lg_dot_wmma(const float* __restrict__ acc, float* __restrict__ out) {
  __shared__ float sd[8][8][32];                    // [wave][vgpr][lane]
  const int lane = threadIdx.x & 31;
  const int wave = threadIdx.x >> 5;
  const int tile = blockIdx.x * 8 + wave;
  const int k0   = tile * 16;
  const int rN   = lane & 15;                       // A row (M) and B column (N)
  const float* U = acc + (size_t)(k0 + rN) * EMBED_DIM;
  const float* I = acc + (size_t)(BATCH + k0 + rN) * EMBED_DIM;

  v8f c = {0.f, 0.f, 0.f, 0.f, 0.f, 0.f, 0.f, 0.f};

#if __has_builtin(__builtin_amdgcn_wmma_f32_16x16x4_f32)
  // f32 path: lanes 0-15 hold K = {col, col+1}; lanes 16-31 hold K = {col+2, col+3}
  const int kb = (lane >> 4) << 1;
  #pragma unroll
  for (int s = 0; s < 16; ++s) {
    const int col = s * 4 + kb;
    v2f a; a[0] = U[col]; a[1] = U[col + 1];
    v2f b; b[0] = I[col]; b[1] = I[col + 1];
    c = __builtin_amdgcn_wmma_f32_16x16x4_f32(false, a, false, b, (short)0, c, false, false);
  }
#else
  // f16 fallback (codegen-confirmed builtin); K=64 done as two K=32 blocks.
  const int kb8 = (lane >> 4) << 3;
  #pragma unroll
  for (int blk = 0; blk < 2; ++blk) {
    v16h a, b;
    #pragma unroll
    for (int i = 0; i < 8; ++i) {
      const int kk = blk * 32 + kb8 + ((i < 4) ? (2 * i) : (16 + 2 * (i - 4)));
      a[2 * i]     = (_Float16)U[kk];
      a[2 * i + 1] = (_Float16)U[kk + 1];
      b[2 * i]     = (_Float16)I[kk];
      b[2 * i + 1] = (_Float16)I[kk + 1];
    }
    c = __builtin_amdgcn_wmma_f32_16x16x32_f16(false, a, false, b, (short)0, c, false, false);
  }
#endif

  #pragma unroll
  for (int g = 0; g < 8; ++g) sd[wave][g][lane] = c[g];
  __syncthreads();

  if (lane < 16) {
    // D diagonal: k<8 -> lane k, VGPR k ; k>=8 -> lane k+16, VGPR k-8
    const float v = (lane < 8) ? sd[wave][lane][lane] : sd[wave][lane - 8][lane + 16];
    out[k0 + lane] = v * 0.0625f;                   // (acc_u/4)·(acc_i/4)
  }
}

extern "C" void kernel_launch(void* const* d_in, const int* in_sizes, int n_in,
                              void* d_out, int out_size, void* d_ws, size_t ws_size,
                              hipStream_t stream) {
  const float* node_emb = (const float*)d_in[0];
  const float* adj_val  = (const float*)d_in[1];
  const int*   adj_row  = (const int*)d_in[2];
  const int*   adj_col  = (const int*)d_in[3];
  const int*   user_ids = (const int*)d_in[4];
  const int*   item_ids = (const int*)d_in[5];
  float* out = (float*)d_out;

  const size_t embElems = (size_t)N_TOTAL * EMBED_DIM;     // 9,606,400 floats
  char* ws   = (char*)d_ws;
  float* embA = (float*)ws;
  float* embB = (float*)(ws + embElems * sizeof(float));
  float* accS = (float*)(ws + 2 * embElems * sizeof(float)); // [2*BATCH][64]

  const dim3 b256(256);
  const int gatherBlocks = (2 * BATCH * EMBED_DIM) / 256;  // 2048
  const int n4 = (int)(embElems / 4);

  lg_gather_init<<<gatherBlocks, b256, 0, stream>>>(node_emb, user_ids, item_ids, accS);

  const float* cur = node_emb;
  float* bufs[2] = {embA, embB};
  for (int L = 0; L < NUM_LAYERS; ++L) {
    float* nxt = bufs[L & 1];
    lg_zero<<<2048, b256, 0, stream>>>((float4*)nxt, n4);
    lg_spmm_scatter<<<2048, b256, 0, stream>>>(cur, adj_val, adj_row, adj_col, nxt);
    lg_gather_acc<<<gatherBlocks, b256, 0, stream>>>(nxt, user_ids, item_ids, accS);
    cur = nxt;
  }

  lg_dot_wmma<<<BATCH / 128, b256, 0, stream>>>(accS, out);  // 32 blocks x 8 waves
}
